// MoE_11390253269332
// MI455X (gfx1250) — matrix operand order repaired and checked by reading, not compiled, not run on hardware
//
#include <hip/hip_runtime.h>
#include <hip/hip_bf16.h>

// ---------------- problem constants (match reference) ----------------
constexpr int Bn = 4096;   // batch
constexpr int Dn = 1024;   // input dim (K of fc1)
constexpr int Hn = 4096;   // hidden dim (N of fc1)
constexpr int En = 8;      // experts
constexpr int LDSK = Dn + 8;               // padded LDS row stride in halves (+16B kills bank conflicts)
constexpr int SH_BYTES = 128 * LDSK * 2;   // 264192 B dynamic LDS (WGP has 320 KB)

// ---------------- WMMA vector types ----------------
typedef __attribute__((ext_vector_type(16))) __bf16 v16bf;
typedef __attribute__((ext_vector_type(8)))  float  v8f;

union FragBF {
    uint4  u[2];
    v16bf  v;
};

__device__ __forceinline__ unsigned short f2bf(float f) {
    unsigned int u = __float_as_uint(f);
    unsigned int r = u + 0x7FFFu + ((u >> 16) & 1u);   // round-to-nearest-even
    return (unsigned short)(r >> 16);
}

// ---------------- kernel 1: x fp32 -> bf16 ----------------
__global__ void convert_x_kernel(const float* __restrict__ x,
                                 unsigned short* __restrict__ xb) {
    size_t i4 = (size_t)blockIdx.x * blockDim.x + threadIdx.x;   // each handles 4 floats
    float4 f = reinterpret_cast<const float4*>(x)[i4];
    uint2 p;
    p.x = (unsigned)f2bf(f.x) | ((unsigned)f2bf(f.y) << 16);
    p.y = (unsigned)f2bf(f.z) | ((unsigned)f2bf(f.w) << 16);
    reinterpret_cast<uint2*>(xb)[i4] = p;
}

// ---------------- kernel 2: w1 [E,D,H] fp32 -> w1t [E,H,D] bf16 ----------------
__global__ void transpose_w1_kernel(const float* __restrict__ w1,
                                    unsigned short* __restrict__ w1t) {
    __shared__ unsigned short tile[32][33];
    const int e  = blockIdx.z;
    const int h0 = blockIdx.x * 32;
    const int d0 = blockIdx.y * 32;
    const int tx = threadIdx.x;   // 0..31
    const int ty = threadIdx.y;   // 0..7
#pragma unroll
    for (int i = 0; i < 4; ++i) {
        int d = d0 + ty + 8 * i;
        tile[ty + 8 * i][tx] = f2bf(w1[((size_t)e * Dn + d) * Hn + h0 + tx]);
    }
    __syncthreads();
#pragma unroll
    for (int i = 0; i < 4; ++i) {
        int h = h0 + ty + 8 * i;
        w1t[((size_t)e * Hn + h) * Dn + d0 + tx] = tile[tx][ty + 8 * i];
    }
}

// ---------------- kernel 3: fused fc1 (WMMA bf16) + bias + ReLU + fc2 dot ----------------
// grid (Bn/128, En), 256 threads = 8 waves. Each wave owns a 16-row M tile;
// sweeps H in groups of 4 N-tiles (4 independent WMMA accumulator chains).
// x tile staged with CDNA5 async global->LDS DMA (ASYNCcnt), no VGPR round-trip.
__global__ __launch_bounds__(256, 1)
void moe_expert_kernel(const unsigned short* __restrict__ xb,   // [B,D] bf16
                       const unsigned short* __restrict__ w1t,  // [E,H,D] bf16
                       const float* __restrict__ b1,            // [E,H]
                       const float* __restrict__ w2,            // [E,H] (C==1)
                       const float* __restrict__ b2,            // [E]
                       float* __restrict__ expert_out) {        // [B,E]
    extern __shared__ unsigned short xs[];                      // 128 rows x LDSK halves
    const int e  = blockIdx.y;
    const int m0 = blockIdx.x * 128;

    // ---- stage x tile (128 x 1024 bf16) into LDS via async DMA, padded rows ----
    {
        const int t = threadIdx.x;
#pragma unroll
        for (int i = 0; i < 64; ++i) {
            int c    = t + 256 * i;      // 16-byte chunk id (8 halves)
            int row  = c >> 7;           // 128 chunks per row
            int col8 = c & 127;
            unsigned int ldsoff =
                (unsigned int)(row * LDSK + col8 * 8) * 2u;   // byte offset in wave's LDS alloc
            unsigned long long gaddr =
                (unsigned long long)(const void*)(xb + (size_t)(m0 + row) * Dn + col8 * 8);
            // CDNA5 async copy: LDS[ldsoff..+15] = MEM[gaddr..+15], tracked by ASYNCcnt
            asm volatile("global_load_async_to_lds_b128 %0, %1, off"
                         :: "v"(ldsoff), "v"(gaddr)
                         : "memory");
        }
        asm volatile("s_wait_asynccnt 0x0" ::: "memory");
    }
    __syncthreads();

    const int wave = threadIdx.x >> 5;
    const int lane = threadIdx.x & 31;
    const int lr   = lane & 15;
    const int half = lane >> 4;

    // A fragment source row in LDS (lanes 0-15 and 16-31 both map to row m0w+lr,
    // but read different K halves per the 16-bit A layout)
    const unsigned short* arow = xs + (wave * 16 + lr) * LDSK;

    float s[8];
#pragma unroll
    for (int r = 0; r < 8; ++r) s[r] = 0.f;

    for (int nb = 0; nb < Hn; nb += 64) {
        v8f acc[4] = {};
        const unsigned short* brow[4];
#pragma unroll
        for (int j = 0; j < 4; ++j) {
            brow[j] = w1t + ((size_t)e * Hn + nb + j * 16 + lr) * Dn;   // lane = column N
            // warm the next panel's stream into GL2 (emits global_prefetch_b8)
            __builtin_prefetch(brow[j] + 64 * Dn, 0, 1);
        }

        for (int k0 = 0; k0 < Dn; k0 += 32) {
            // A 16x32 bf16: lanes 0-15 hold K {0..7, 16..23}; lanes 16-31 hold K {8..15, 24..31}
            FragBF a;
            a.u[0] = *reinterpret_cast<const uint4*>(arow + k0 + half * 8);
            a.u[1] = *reinterpret_cast<const uint4*>(arow + k0 + 16 + half * 8);
#pragma unroll
            for (int j = 0; j < 4; ++j) {
                // B 32x16 bf16: lane lr = column N, K {half*8 + 0..7, 16 + half*8 + 0..7} (matches A)
                FragBF bfr;
                bfr.u[0] = *reinterpret_cast<const uint4*>(brow[j] + k0 + half * 8);
                bfr.u[1] = *reinterpret_cast<const uint4*>(brow[j] + k0 + 16 + half * 8);
                acc[j] = __builtin_amdgcn_wmma_f32_16x16x32_bf16(
                    /*neg_a=*/false, a.v, /*neg_b=*/false, bfr.v,
                    /*c_mod=*/(short)0, acc[j], /*reuse_a=*/false, /*reuse_b=*/false);
            }
        }
        // epilogue: h = relu(acc + b1); s += h * w2col  (fc2 with C==1 fused)
#pragma unroll
        for (int j = 0; j < 4; ++j) {
            const int n = nb + j * 16 + lr;            // C/D layout: lane's N = lr
            const float b1v = b1[(size_t)e * Hn + n];
            const float w2v = w2[(size_t)e * Hn + n];
#pragma unroll
            for (int r = 0; r < 8; ++r) {
                float hv = acc[j][r] + b1v;
                s[r] += fmaxf(hv, 0.f) * w2v;
            }
        }
    }

    // C/D layout: VGPR r -> M = half*8 + r; reduce the 16 N-partials across each 16-lane group
    const float b2v = b2[e];
#pragma unroll
    for (int r = 0; r < 8; ++r) {
        float v = s[r];
        v += __shfl_xor(v, 1, 32);
        v += __shfl_xor(v, 2, 32);
        v += __shfl_xor(v, 4, 32);
        v += __shfl_xor(v, 8, 32);
        if (lr == 0) {
            int mrow = m0 + wave * 16 + half * 8 + r;
            expert_out[(size_t)mrow * En + e] = v + b2v;
        }
    }
}

// ---------------- kernel 4: gate softmax + mixture ----------------
__global__ __launch_bounds__(256)
void gate_mixture_kernel(const float* __restrict__ x,          // [B,D] fp32
                         const float* __restrict__ wg,         // [D,8]
                         const float* __restrict__ bg,         // [8]
                         const float* __restrict__ expert_out, // [B,8]
                         float* __restrict__ out) {            // [B] mixture ++ [B,8] gate
    const int wave = threadIdx.x >> 5;
    const int lane = threadIdx.x & 31;
    const int b = blockIdx.x * 8 + wave;
    const float* xrow = x + (size_t)b * Dn;
    float acc[8] = {};
    for (int d = lane; d < Dn; d += 32) {
        const float xv = xrow[d];
        const float* wr = wg + (size_t)d * 8;
#pragma unroll
        for (int j = 0; j < 8; ++j) acc[j] = fmaf(xv, wr[j], acc[j]);
    }
#pragma unroll
    for (int j = 0; j < 8; ++j) {
        acc[j] += __shfl_xor(acc[j], 16, 32);
        acc[j] += __shfl_xor(acc[j], 8, 32);
        acc[j] += __shfl_xor(acc[j], 4, 32);
        acc[j] += __shfl_xor(acc[j], 2, 32);
        acc[j] += __shfl_xor(acc[j], 1, 32);
    }
    if (lane == 0) {
        float l[8], mx = -3.4e38f;
#pragma unroll
        for (int j = 0; j < 8; ++j) { l[j] = acc[j] + bg[j]; mx = fmaxf(mx, l[j]); }
        float g[8], sum = 0.f;
#pragma unroll
        for (int j = 0; j < 8; ++j) { g[j] = __expf(l[j] - mx); sum += g[j]; }
        const float inv = 1.f / sum;
        float mix = 0.f;
#pragma unroll
        for (int j = 0; j < 8; ++j) {
            g[j] *= inv;
            out[(size_t)Bn + (size_t)b * 8 + j] = g[j];
            mix = fmaf(g[j], expert_out[(size_t)b * 8 + j], mix);
        }
        out[b] = mix;
    }
}

// ---------------- launch ----------------
extern "C" void kernel_launch(void* const* d_in, const int* in_sizes, int n_in,
                              void* d_out, int out_size, void* d_ws, size_t ws_size,
                              hipStream_t stream) {
    const float* x  = (const float*)d_in[0];   // [B,D]
    const float* w1 = (const float*)d_in[1];   // [E,D,H]
    const float* b1 = (const float*)d_in[2];   // [E,H]
    const float* w2 = (const float*)d_in[3];   // [E,H,1]
    const float* b2 = (const float*)d_in[4];   // [E,1]
    const float* wg = (const float*)d_in[5];   // [D,8]
    const float* bg = (const float*)d_in[6];   // [8]
    float* out = (float*)d_out;

    // workspace layout (~72.1 MB): xb (8 MB) | w1t (64 MB) | expert_out (128 KB)
    char* ws = (char*)d_ws;
    const size_t XB_BYTES  = (size_t)Bn * Dn * 2;
    const size_t W1T_BYTES = (size_t)En * Hn * Dn * 2;
    unsigned short* xb   = (unsigned short*)ws;
    unsigned short* w1t  = (unsigned short*)(ws + XB_BYTES);
    float*          eout = (float*)(ws + XB_BYTES + W1T_BYTES);

    (void)in_sizes; (void)n_in; (void)out_size; (void)ws_size;

    // allow >64KB dynamic LDS for the fused GEMM kernel (idempotent, capture-safe)
    (void)hipFuncSetAttribute((const void*)moe_expert_kernel,
                              hipFuncAttributeMaxDynamicSharedMemorySize, SH_BYTES);

    convert_x_kernel<<<(Bn * Dn) / (256 * 4), 256, 0, stream>>>(x, xb);
    transpose_w1_kernel<<<dim3(Hn / 32, Dn / 32, En), dim3(32, 8), 0, stream>>>(w1, w1t);
    moe_expert_kernel<<<dim3(Bn / 128, En), 256, SH_BYTES, stream>>>(xb, w1t, b1, w2, b2, eout);
    gate_mixture_kernel<<<Bn / 8, 256, 0, stream>>>(x, wg, bg, eout, out);
}
